// UncGuidedModulatedDCNBlock_67250597921160
// MI455X (gfx1250) — matrix operand order repaired and checked by reading, not compiled, not run on hardware
//
#include <hip/hip_runtime.h>
#include <stdint.h>

// Problem constants (match reference)
#define BB   4
#define CC   128
#define HH   128
#define WW   128
#define KT   9          // 3x3 taps
#define HW   (HH*WW)
#define LDA  40         // LDS row stride (bf16 elems): 32 + pad -> 80B rows, 16B aligned
#define LDB  40
#define NCHUNK (KT * 4) // 36 K-chunks of 32

typedef __attribute__((ext_vector_type(16))) __bf16 v16bf;
typedef __attribute__((ext_vector_type(8)))  __bf16 v8bf;
typedef __attribute__((ext_vector_type(8)))  float  v8f;

__device__ __forceinline__ unsigned f2bf(float f) {
  union { float f; unsigned u; } v; v.f = f;
  unsigned r = v.u + 0x7FFFu + ((v.u >> 16) & 1u);  // round-to-nearest-even
  return r >> 16;
}
__device__ __forceinline__ unsigned pk2bf(float a, float b) {
  return f2bf(a) | (f2bf(b) << 16);
}
__device__ __forceinline__ float sigm(float x) { return 1.0f / (1.0f + __expf(-x)); }

// Async global->LDS copy of 32 bytes (two b128), tracked by ASYNCcnt.
// IOFFSET is applied to BOTH the LDS and global addresses (ISA 15.18.3).
__device__ __forceinline__ void async_copy_b256(void* lds_dst, const void* gsrc) {
  unsigned lb = (unsigned)(size_t)lds_dst;     // LDS aperture: addr[31:0] == LDS offset
  asm volatile(
      "global_load_async_to_lds_b128 %0, %1, off\n\t"
      "global_load_async_to_lds_b128 %0, %1, off offset:16"
      :: "v"(lb), "v"(gsrc) : "memory");
}
__device__ __forceinline__ void wait_async0() {
  asm volatile("s_wait_asynccnt 0x0" ::: "memory");
}

// ---------------------------------------------------------------------------
// Weight pre-transform: w[n][c][ky][kx] (f32, OIHW) -> wT[tap][n][c] (bf16)
// ---------------------------------------------------------------------------
__global__ void prep_w_kernel(const float* __restrict__ w, unsigned short* __restrict__ wT) {
  int idx = blockIdx.x * 256 + threadIdx.x;
  if (idx >= KT * CC * CC) return;
  int kk = idx / (CC * CC);
  int r  = idx - kk * (CC * CC);
  int n  = r >> 7;
  int c  = r & 127;
  wT[(kk * CC + n) * CC + c] = (unsigned short)f2bf(w[(n * CC + c) * KT + kk]);
}

// ---------------------------------------------------------------------------
// Offset/mask conv (27 out-ch over 129 in-ch, 3x3) + gate.  One thread/pixel.
// ---------------------------------------------------------------------------
__global__ __launch_bounds__(256) void om_gate_kernel(
    const float* __restrict__ x, const float* __restrict__ U,
    const float* __restrict__ w_om, const float* __restrict__ b_om,
    float* __restrict__ offs, float* __restrict__ mask, float* __restrict__ gate) {
  int pid = blockIdx.x * 256 + threadIdx.x;   // exactly B*H*W threads
  int b  = pid >> 14;
  int y  = (pid >> 7) & 127;
  int xm = pid & 127;

  float acc[27];
#pragma unroll
  for (int o = 0; o < 27; ++o) acc[o] = b_om[o];

  for (int c = 0; c < CC + 1; ++c) {
#pragma unroll
    for (int ky = 0; ky < 3; ++ky) {
      int yy = y + ky - 1;
      bool yok = (yy >= 0) && (yy < HH);
#pragma unroll
      for (int kx = 0; kx < 3; ++kx) {
        int xx = xm + kx - 1;
        float gv = 0.0f;
        if (yok && xx >= 0 && xx < WW) {
          if (c < CC) gv = x[((b * CC + c) * HH + yy) * WW + xx];
          else {
            float u = U[b * HW + yy * WW + xx];
            gv = fminf(fmaxf(u, 0.0f), 1.0f);
          }
        }
        const float* wp = w_om + (c * 3 + ky) * 3 + kx;  // stride per out-ch = 129*9
#pragma unroll
        for (int o = 0; o < 27; ++o)
          acc[o] = fmaf(gv, wp[o * (CC + 1) * 9], acc[o]);
      }
    }
  }

  int pix = y * WW + xm;
#pragma unroll
  for (int ch = 0; ch < 18; ++ch) offs[(b * 18 + ch) * HW + pix] = acc[ch];
#pragma unroll
  for (int kk = 0; kk < 9; ++kk)  mask[(b * 9 + kk) * HW + pix] = 2.0f * sigm(acc[18 + kk]);
  float u = fminf(fmaxf(U[pid], 0.0f), 1.0f);
  gate[pid] = sigm(10.0f * (u - 0.5f));     // sigmoid(BETA*(U-TAU))
}

// ---------------------------------------------------------------------------
// WMMA fragment helpers (documented gfx1250 16-bit layouts)
// ---------------------------------------------------------------------------
__device__ __forceinline__ v16bf load_frag_A(const unsigned short* Als, int lane, int mW) {
  int m  = mW + (lane & 15);
  int kb = (lane < 16) ? 0 : 8;
  v8bf lo = *(const v8bf*)&Als[m * LDA + kb];
  v8bf hi = *(const v8bf*)&Als[m * LDA + kb + 16];
  return __builtin_shufflevector(lo, hi, 0,1,2,3,4,5,6,7,8,9,10,11,12,13,14,15);
}
__device__ __forceinline__ v16bf load_frag_B(const unsigned short* Bls, int lane, int n0) {
  int n  = n0 + (lane & 15);
  int kb = (lane < 16) ? 0 : 8;
  v8bf lo = *(const v8bf*)&Bls[n * LDB + kb];
  v8bf hi = *(const v8bf*)&Bls[n * LDB + kb + 16];
  return __builtin_shufflevector(lo, hi, 0,1,2,3,4,5,6,7,8,9,10,11,12,13,14,15);
}

// Register staging for one K-chunk (per thread): 16 bf16 of A.
struct Stage {
  unsigned a[8];
};

__device__ __forceinline__ void commit_stage(unsigned short* __restrict__ Als,
                                             int mA, int khA, const Stage& s) {
  uint4 lo, hi;
  lo.x = s.a[0]; lo.y = s.a[1]; lo.z = s.a[2]; lo.w = s.a[3];
  hi.x = s.a[4]; hi.y = s.a[5]; hi.z = s.a[6]; hi.w = s.a[7];
  *(uint4*)&Als[mA * LDA + khA]     = lo;
  *(uint4*)&Als[mA * LDA + khA + 8] = hi;
}

// ---------------------------------------------------------------------------
// Implicit-GEMM conv3x3 (C=128 -> C=128), bf16-in, double-buffered pipeline,
// async B-tile copies (global -> LDS, ASYNCcnt).
// MODE 0: out(bf16) = relu(conv(in) + bias)
// MODE 1: out(f32)  = relu(resx + (conv(in) + bias) * gate)
// ---------------------------------------------------------------------------
template <int MODE>
__global__ __launch_bounds__(256) void gemm_conv_kernel(
    const unsigned short* __restrict__ in, const unsigned short* __restrict__ wT,
    const float* __restrict__ bias, const float* __restrict__ resx,
    const float* __restrict__ gate, void* __restrict__ outv) {
  __shared__ alignas(16) unsigned short Als[2][128 * LDA];
  __shared__ alignas(16) unsigned short Bls[2][128 * LDB];

  int t    = threadIdx.x;
  int lane = t & 31;
  int wv   = t >> 5;
  int row  = blockIdx.x;            // b*HH + y
  int b    = row >> 7;
  int y    = row & 127;
  int mB   = (wv & 3) * 32;         // wave's 2 M sub-tiles at mB, mB+16
  int nG   = (wv >> 2) * 64;        // wave's N group (4 sub-tiles)

  v8f acc[2][4];
#pragma unroll
  for (int r = 0; r < 2; ++r)
#pragma unroll
    for (int s = 0; s < 4; ++s)
#pragma unroll
      for (int j = 0; j < 8; ++j) acc[r][s][j] = 0.0f;

  int mA  = t & 127;                // pixel this thread stages
  int khA = (t >> 7) * 16;          // 16-channel half of the K-chunk
  int nB  = t >> 1;                 // B-tile row
  int hB  = (t & 1) * 16;           // bf16 element offset within B row

  // Async B-tile copy for chunk ch into buffer pp
  auto launch_B = [&](int ch, int pp) {
    int kk = ch >> 2;
    int c0 = (ch & 3) << 5;
    async_copy_b256(&Bls[pp][nB * LDB + hB], wT + (kk * CC + nB) * CC + c0 + hB);
  };

  // Per-thread A staging for chunk ch: pixel mA, channels c0+khA..+15, bf16 copy
  auto stage_chunk = [&](int ch, Stage& s) {
    int kk = ch >> 2;
    int c0 = (ch & 3) << 5;
    int dy = kk / 3 - 1, dx = kk % 3 - 1;
    int yy = y + dy;
    int xx = mA + dx;
    bool ok = (yy >= 0) && (yy < HH) && (xx >= 0) && (xx < WW);
    if (ok) {
      const unsigned short* src =
          in + (((size_t)(b * CC + c0 + khA) * HH + yy) * WW + xx);
#pragma unroll
      for (int i = 0; i < 8; ++i) {
        unsigned lo = src[(2 * i) * HW];
        unsigned hi = src[(2 * i + 1) * HW];
        s.a[i] = lo | (hi << 16);
      }
    } else {
#pragma unroll
      for (int i = 0; i < 8; ++i) s.a[i] = 0u;
    }
  };

  // --- Pipeline prologue
  Stage sreg;
  launch_B(0, 0);
  stage_chunk(0, sreg);
  commit_stage(Als[0], mA, khA, sreg);
  wait_async0();
  __syncthreads();

  // --- Main pipelined loop: one barrier per chunk
  for (int ch = 0; ch < NCHUNK; ++ch) {
    int p = ch & 1;
    bool more = (ch + 1 < NCHUNK);
    Stage snext;
    if (more) {
      launch_B(ch + 1, p ^ 1);
      stage_chunk(ch + 1, snext);
    }

    v16bf a0 = load_frag_A(Als[p], lane, mB);
    v16bf a1 = load_frag_A(Als[p], lane, mB + 16);
#pragma unroll
    for (int s = 0; s < 4; ++s) {
      v16bf bfrag = load_frag_B(Bls[p], lane, nG + s * 16);
      acc[0][s] = __builtin_amdgcn_wmma_f32_16x16x32_bf16(
          false, a0, false, bfrag, (short)0, acc[0][s], false, false);
      acc[1][s] = __builtin_amdgcn_wmma_f32_16x16x32_bf16(
          false, a1, false, bfrag, (short)0, acc[1][s], false, false);
    }

    if (more) {
      commit_stage(Als[p ^ 1], mA, khA, snext);
      wait_async0();
      __syncthreads();
    }
  }

  // --- Epilogue: C/D layout — VGPR j holds (M=j | j+8), N=lane&15
  int ln = lane & 15;
#pragma unroll
  for (int r = 0; r < 2; ++r) {
    int mo = mB + r * 16 + ((lane >> 4) << 3);
#pragma unroll
    for (int s = 0; s < 4; ++s) {
      int n = nG + s * 16 + ln;
      float bv = bias[n];
#pragma unroll
      for (int j = 0; j < 8; ++j) {
        int xx = mo + j;
        long o = ((long)(b * CC + n) * HH + y) * WW + xx;
        float v = acc[r][s][j] + bv;
        if (MODE == 0) {
          v = fmaxf(v, 0.0f);
          ((unsigned short*)outv)[o] = (unsigned short)f2bf(v);
        } else {
          float g = gate[(long)b * HW + y * WW + xx];
          ((float*)outv)[o] = fmaxf(resx[o] + v * g, 0.0f);
        }
      }
    }
  }
}

// ---------------------------------------------------------------------------
// Modulated DCN as implicit GEMM (A = bilinear(x)*mask, bf16), pipelined,
// async B-tile copies.  Writes z as bf16 for the next conv stage.
// ---------------------------------------------------------------------------
__global__ __launch_bounds__(256) void gemm_dcn_kernel(
    const float* __restrict__ x, const unsigned short* __restrict__ wT,
    const float* __restrict__ bias, const float* __restrict__ offs,
    const float* __restrict__ mask, unsigned short* __restrict__ out) {
  __shared__ alignas(16) unsigned short Als[2][128 * LDA];
  __shared__ alignas(16) unsigned short Bls[2][128 * LDB];
  __shared__ int   cIdx[2][128][4];
  __shared__ float cWgt[2][128][4];

  int t    = threadIdx.x;
  int lane = t & 31;
  int wv   = t >> 5;
  int row  = blockIdx.x;            // b*HH + y
  int b    = row >> 7;
  int y    = row & 127;
  int mB   = (wv & 3) * 32;
  int nG   = (wv >> 2) * 64;

  v8f acc[2][4];
#pragma unroll
  for (int r = 0; r < 2; ++r)
#pragma unroll
    for (int s = 0; s < 4; ++s)
#pragma unroll
      for (int j = 0; j < 8; ++j) acc[r][s][j] = 0.0f;

  int mA  = t & 127;
  int khA = (t >> 7) * 16;
  int nB  = t >> 1;
  int hB  = (t & 1) * 16;

  auto launch_B = [&](int ch, int pp) {
    int kk = ch >> 2;
    int c0 = (ch & 3) << 5;
    async_copy_b256(&Bls[pp][nB * LDB + hB], wT + (kk * CC + nB) * CC + c0 + hB);
  };

  // Bilinear corner setup for tap kk into buffer q (threads 0..127)
  auto make_corners = [&](int kk, int q) {
    if (t < 128) {
      int m   = t;
      int pix = y * WW + m;
      float dyv = offs[(b * 18 + 2 * kk    ) * HW + pix];
      float dxv = offs[(b * 18 + 2 * kk + 1) * HW + pix];
      float mk  = mask[(b * 9 + kk) * HW + pix];
      float py = (float)y + (float)(kk / 3 - 1) + dyv;
      float px = (float)m + (float)(kk % 3 - 1) + dxv;
      float fy = floorf(py), fx = floorf(px);
      float ly = py - fy, lx = px - fx;
      float wy[2] = {1.0f - ly, ly};
      float wx[2] = {1.0f - lx, lx};
      float ys[2] = {fy, fy + 1.0f};
      float xs[2] = {fx, fx + 1.0f};
#pragma unroll
      for (int i = 0; i < 4; ++i) {
        float yc = ys[i >> 1], xc = xs[i & 1];
        bool valid = (yc >= 0.0f) && (yc <= (float)(HH - 1)) &&
                     (xc >= 0.0f) && (xc <= (float)(WW - 1));
        int yi = (int)fminf(fmaxf(yc, 0.0f), (float)(HH - 1));
        int xi = (int)fminf(fmaxf(xc, 0.0f), (float)(WW - 1));
        cIdx[q][m][i] = yi * WW + xi;
        cWgt[q][m][i] = valid ? (wy[i >> 1] * wx[i & 1] * mk) : 0.0f;
      }
    }
  };

  // Stage one chunk: 16 channels of modulated bilinear samples for pixel mA
  auto stage_chunk = [&](int ch, Stage& s) {
    int kk = ch >> 2;
    int c0 = (ch & 3) << 5;
    int q  = kk & 1;
    int   i0 = cIdx[q][mA][0], i1 = cIdx[q][mA][1];
    int   i2 = cIdx[q][mA][2], i3 = cIdx[q][mA][3];
    float w0 = cWgt[q][mA][0], w1 = cWgt[q][mA][1];
    float w2 = cWgt[q][mA][2], w3 = cWgt[q][mA][3];
    const float* plane = x + (size_t)(b * CC + c0 + khA) * HW;
#pragma unroll
    for (int i = 0; i < 8; ++i) {
      const float* p0 = plane + (size_t)(2 * i) * HW;
      const float* p1 = plane + (size_t)(2 * i + 1) * HW;
      float va = w0 * p0[i0] + w1 * p0[i1] + w2 * p0[i2] + w3 * p0[i3];
      float vb = w0 * p1[i0] + w1 * p1[i1] + w2 * p1[i2] + w3 * p1[i3];
      s.a[i] = pk2bf(va, vb);
    }
  };

  // --- Prologue: corners for tap 0, then stage+commit chunk 0
  make_corners(0, 0);
  __syncthreads();
  Stage sreg;
  launch_B(0, 0);
  stage_chunk(0, sreg);
  commit_stage(Als[0], mA, khA, sreg);
  wait_async0();
  __syncthreads();

  // --- Main pipelined loop
  for (int ch = 0; ch < NCHUNK; ++ch) {
    int p = ch & 1;
    bool more = (ch + 1 < NCHUNK);
    Stage snext;
    if (more) {
      launch_B(ch + 1, p ^ 1);
      stage_chunk(ch + 1, snext);
    }

    v16bf a0 = load_frag_A(Als[p], lane, mB);
    v16bf a1 = load_frag_A(Als[p], lane, mB + 16);
#pragma unroll
    for (int s = 0; s < 4; ++s) {
      v16bf bfrag = load_frag_B(Bls[p], lane, nG + s * 16);
      acc[0][s] = __builtin_amdgcn_wmma_f32_16x16x32_bf16(
          false, a0, false, bfrag, (short)0, acc[0][s], false, false);
      acc[1][s] = __builtin_amdgcn_wmma_f32_16x16x32_bf16(
          false, a1, false, bfrag, (short)0, acc[1][s], false, false);
    }

    // Compute next tap's corners one tap ahead (3rd chunk of current tap),
    // into the other corner buffer; visible after this iteration's barrier.
    if ((ch & 3) == 2) {
      int kn = (ch >> 2) + 1;
      if (kn < KT) make_corners(kn, kn & 1);
    }

    if (more) {
      commit_stage(Als[p ^ 1], mA, khA, snext);
      wait_async0();
      __syncthreads();
    }
  }

  // --- Epilogue: write z as bf16
  int ln = lane & 15;
#pragma unroll
  for (int r = 0; r < 2; ++r) {
    int mo = mB + r * 16 + ((lane >> 4) << 3);
#pragma unroll
    for (int s = 0; s < 4; ++s) {
      int n = nG + s * 16 + ln;
      float bv = bias[n];
#pragma unroll
      for (int j = 0; j < 8; ++j) {
        long o = ((long)(b * CC + n) * HH + y) * WW + (mo + j);
        out[o] = (unsigned short)f2bf(acc[r][s][j] + bv);
      }
    }
  }
}

// ---------------------------------------------------------------------------
// Host-side launcher
// ---------------------------------------------------------------------------
extern "C" void kernel_launch(void* const* d_in, const int* in_sizes, int n_in,
                              void* d_out, int out_size, void* d_ws, size_t ws_size,
                              hipStream_t stream) {
  const float* x     = (const float*)d_in[0];
  const float* U     = (const float*)d_in[1];
  const float* w_om  = (const float*)d_in[2];
  const float* b_om  = (const float*)d_in[3];
  const float* w_dcn = (const float*)d_in[4];
  const float* b_dcn = (const float*)d_in[5];
  const float* w1    = (const float*)d_in[6];
  const float* b1    = (const float*)d_in[7];
  const float* w2    = (const float*)d_in[8];
  const float* b2    = (const float*)d_in[9];
  float* out = (float*)d_out;

  // Carve workspace
  char* p = (char*)d_ws;
  auto carve = [&](size_t sz) { void* r = (void*)p; p += (sz + 255) & ~(size_t)255; return r; };
  unsigned short* wT_dcn = (unsigned short*)carve((size_t)KT * CC * CC * 2);
  unsigned short* wT_1   = (unsigned short*)carve((size_t)KT * CC * CC * 2);
  unsigned short* wT_2   = (unsigned short*)carve((size_t)KT * CC * CC * 2);
  float* gate = (float*)carve((size_t)BB * HW * 4);
  float* offs = (float*)carve((size_t)BB * 18 * HW * 4);
  float* mask = (float*)carve((size_t)BB * 9 * HW * 4);
  unsigned short* z  = (unsigned short*)carve((size_t)BB * CC * HW * 2);  // bf16
  unsigned short* r1 = (unsigned short*)carve((size_t)BB * CC * HW * 2);  // bf16

  const int WPREP_B = (KT * CC * CC + 255) / 256;
  prep_w_kernel<<<WPREP_B, 256, 0, stream>>>(w_dcn, wT_dcn);
  prep_w_kernel<<<WPREP_B, 256, 0, stream>>>(w1, wT_1);
  prep_w_kernel<<<WPREP_B, 256, 0, stream>>>(w2, wT_2);

  om_gate_kernel<<<(BB * HW) / 256, 256, 0, stream>>>(x, U, w_om, b_om, offs, mask, gate);

  const int GEMM_B = BB * HH;   // 512 blocks: one full image row each
  gemm_dcn_kernel<<<GEMM_B, 256, 0, stream>>>(x, wT_dcn, b_dcn, offs, mask, z);
  gemm_conv_kernel<0><<<GEMM_B, 256, 0, stream>>>(z, wT_1, b1, x, gate, (void*)r1);
  gemm_conv_kernel<1><<<GEMM_B, 256, 0, stream>>>(r1, wT_2, b2, x, gate, (void*)out);
}
